// HyperGraphAttentionOutLayer_28673201668376
// MI455X (gfx1250) — matrix operand order repaired and checked by reading, not compiled, not run on hardware
//
#include <hip/hip_runtime.h>
#include <hip/hip_bf16.h>

typedef __attribute__((ext_vector_type(16))) __bf16 v16bf;
typedef __attribute__((ext_vector_type(8)))  float  v8f;
typedef __attribute__((ext_vector_type(4)))  float  f32x4;

union FragAB { v16bf v; f32x4 q[2]; };

static constexpr int   B_ = 4;
static constexpr int   N_ = 2048;     // N1 == N2
static constexpr int   F_ = 256;      // F == O
static constexpr float kAlpha  = 0.2f;
static constexpr float kNegInf = -9000000000000000.0f;

// ---------------------------------------------------------------------------
// K1: v1 = W1 @ w3[:O], v2 = W2 @ w3[O:]. One block of 256 threads (f = tid).
// ---------------------------------------------------------------------------
__global__ void k_prep_v(const float* __restrict__ W1, const float* __restrict__ W2,
                         const float* __restrict__ w3,
                         float* __restrict__ v1, float* __restrict__ v2) {
  int f = threadIdx.x;
  const float* r1 = W1 + f * F_;
  const float* r2 = W2 + f * F_;
  float a1 = 0.f, a2 = 0.f;
  for (int o = 0; o < F_; ++o) {
    a1 += r1[o] * w3[o];
    a2 += r2[o] * w3[F_ + o];
  }
  v1[f] = a1;
  v2[f] = a2;
}

// ---------------------------------------------------------------------------
// K2: per j-tile of 32 rows: s1[b,j]=word·v1, t2[b,j]=word·v2, and write the
// bf16 transposed copy wT[b][f][j] (coalesced via an LDS tile transpose).
// ---------------------------------------------------------------------------
__global__ void k_word_pre(const float* __restrict__ word,
                           const float* __restrict__ v1, const float* __restrict__ v2,
                           float* __restrict__ s1, float* __restrict__ t2,
                           __bf16* __restrict__ wT) {
  __shared__ __bf16 tileT[F_][34];      // [f][j], padded against bank conflicts
  __shared__ float  part1[32][8];
  __shared__ float  part2[32][8];

  const int tid  = threadIdx.x;
  const int lane = tid & 31;
  const int wave = tid >> 5;
  const int b    = blockIdx.x >> 6;
  const int j0   = (blockIdx.x & 63) * 32;

  const float va = v1[tid];             // this thread owns feature f = tid
  const float vb = v2[tid];

  for (int r = 0; r < 32; ++r) {
    float x = word[(((size_t)b * N_) + j0 + r) * F_ + tid];
    tileT[tid][r] = (__bf16)x;
    float p1 = x * va;
    float p2 = x * vb;
    for (int off = 16; off; off >>= 1) {
      p1 += __shfl_xor(p1, off, 32);
      p2 += __shfl_xor(p2, off, 32);
    }
    if (lane == 0) { part1[r][wave] = p1; part2[r][wave] = p2; }
  }
  __syncthreads();

  if (tid < 32) {
    float a1 = 0.f, a2 = 0.f;
    for (int w = 0; w < 8; ++w) { a1 += part1[tid][w]; a2 += part2[tid][w]; }
    s1[(size_t)b * N_ + j0 + tid] = a1;
    t2[(size_t)b * N_ + j0 + tid] = a2;
  }

  // coalesced bf16 write-out of the transposed tile
  for (int idx = tid; idx < F_ * 32; idx += 256) {
    int f = idx >> 5, j = idx & 31;
    wT[((size_t)(b * F_ + f)) * N_ + j0 + j] = tileT[f][j];
  }
}

// ---------------------------------------------------------------------------
// K3: s2[b,i] = (adj[b,i,:]·t2[b,:]) / max(deg,1). One wave per adj row.
// ---------------------------------------------------------------------------
__global__ void k_s2(const float* __restrict__ adj, const float* __restrict__ t2,
                     float* __restrict__ s2) {
  const int tid  = threadIdx.x;
  const int lane = tid & 31;
  const int wave = tid >> 5;
  const int row  = blockIdx.x * 8 + wave;   // 0 .. B_*N_-1
  const int b    = row >> 11;

  const float* ar  = adj + (size_t)row * N_;
  const float* t2b = t2 + (size_t)b * N_;
  float deg = 0.f, acc = 0.f;
  for (int k = 0; k < N_ / 32; ++k) {
    int j = lane + 32 * k;
    float a = ar[j];
    deg += a;
    acc += a * t2b[j];
  }
  for (int off = 16; off; off >>= 1) {
    deg += __shfl_xor(deg, off, 32);
    acc += __shfl_xor(acc, off, 32);
  }
  if (lane == 0) s2[row] = acc / (deg > 0.f ? deg : 1.f);
}

// ---------------------------------------------------------------------------
// K4: fused attention softmax + edge = softmax(attn) @ word via WMMA bf16.
// Block = (b, 16-row tile). 8 waves; wave w owns output cols [32w, 32w+32).
// ---------------------------------------------------------------------------
__global__ void __launch_bounds__(256)
k_attn_edge(const float* __restrict__ adj, const float* __restrict__ s1g,
            const float* __restrict__ s2g, const __bf16* __restrict__ wT,
            float* __restrict__ out) {
  __shared__ __bf16 pch[16][264];       // bf16 softmax numerators, 256-j chunk
  __shared__ float  partred[16][8];
  __shared__ float  rowmaxS[16];
  __shared__ float  rowinvS[16];
  __shared__ float  s2S[16];

  const int tid  = threadIdx.x;
  const int lane = tid & 31;
  const int wave = tid >> 5;
  const int b    = blockIdx.x >> 7;
  const int i0   = (blockIdx.x & 127) * 16;

  if (tid < 16) s2S[tid] = s2g[b * N_ + i0 + tid];
  __syncthreads();

  const float* s1b  = s1g + (size_t)b * N_;
  const float* adjb = adj + ((size_t)(b * N_ + i0)) * N_;

  // ---- Pass 1: per-row maxima of attention = (adj>0 ? exp(lrelu) : -inf) ----
  for (int r = 0; r < 16; ++r) {
    const float* arow = adjb + (size_t)r * N_;
    const float  s2v  = s2S[r];
    float mm = kNegInf;
    for (int k = 0; k < 8; ++k) {
      int   j   = tid + 256 * k;
      float a   = arow[j];
      float pre = a * (s2v + s1b[j]);
      float lr  = pre > 0.f ? pre : kAlpha * pre;
      float att = (a > 0.f) ? __expf(lr) : kNegInf;
      mm = fmaxf(mm, att);
    }
    for (int off = 16; off; off >>= 1) mm = fmaxf(mm, __shfl_xor(mm, off, 32));
    if (lane == 0) partred[r][wave] = mm;
  }
  __syncthreads();
  if (tid < 16) {
    float mm = kNegInf;
    for (int w = 0; w < 8; ++w) mm = fmaxf(mm, partred[tid][w]);
    rowmaxS[tid] = mm;
  }
  __syncthreads();

  // ---- Pass 2: chunked softmax numerators (bf16) + WMMA accumulation ----
  v8f acc0 = {0.f, 0.f, 0.f, 0.f, 0.f, 0.f, 0.f, 0.f};
  v8f acc1 = {0.f, 0.f, 0.f, 0.f, 0.f, 0.f, 0.f, 0.f};
  float psum[16];
#pragma unroll
  for (int r = 0; r < 16; ++r) psum[r] = 0.f;

  const int arow16 = lane & 15;               // A-matrix M row this lane serves
  const int aoff   = (lane < 16) ? 0 : 8;     // A K-offset (per wave32 layout)
  const int khalf  = (lane < 16) ? 0 : 8;     // B K-offset (matches A)
  const __bf16* wTb = wT + (size_t)b * F_ * N_;

  for (int jb = 0; jb < 8; ++jb) {
    __syncthreads();                          // pch reuse barrier
    const int jbase = jb * 256;
#pragma unroll
    for (int r = 0; r < 16; ++r) {
      int   j   = jbase + tid;
      float a   = adjb[(size_t)r * N_ + j];
      float pre = a * (s2S[r] + s1b[j]);
      float lr  = pre > 0.f ? pre : kAlpha * pre;
      float att = (a > 0.f) ? __expf(lr) : kNegInf;
      float e   = __expf(att - rowmaxS[r]);   // in [0,1]; 0 for masked entries
      pch[r][tid] = (__bf16)e;
      psum[r] += e;
    }
    __syncthreads();

#pragma unroll
    for (int jc = 0; jc < 8; ++jc) {
      const int j0l = jc * 32;
      FragAB A;
      A.q[0] = *(const f32x4*)&pch[arow16][j0l + aoff];
      A.q[1] = *(const f32x4*)&pch[arow16][j0l + aoff + 16];

      const int jg = jbase + j0l + khalf;
      FragAB B0, B1;
      const __bf16* p0 = wTb + (size_t)(wave * 32 + arow16) * N_ + jg;
      const __bf16* p1 = wTb + (size_t)(wave * 32 + 16 + arow16) * N_ + jg;
      B0.q[0] = *(const f32x4*)p0;
      B0.q[1] = *(const f32x4*)(p0 + 16);
      B1.q[0] = *(const f32x4*)p1;
      B1.q[1] = *(const f32x4*)(p1 + 16);

      acc0 = __builtin_amdgcn_wmma_f32_16x16x32_bf16(false, A.v, false, B0.v,
                                                     (short)0, acc0, false, false);
      acc1 = __builtin_amdgcn_wmma_f32_16x16x32_bf16(false, A.v, false, B1.v,
                                                     (short)0, acc1, false, false);
    }
  }

  // ---- Row-sum reduction -> 1/sum folded into the store ----
#pragma unroll
  for (int r = 0; r < 16; ++r) {
    float s = psum[r];
    for (int off = 16; off; off >>= 1) s += __shfl_xor(s, off, 32);
    if (lane == 0) partred[r][wave] = s;
  }
  __syncthreads();
  if (tid < 16) {
    float s = 0.f;
    for (int w = 0; w < 8; ++w) s += partred[tid][w];
    rowinvS[tid] = 1.f / s;
  }
  __syncthreads();

  // ---- Store: D VGPR r maps to (M = r + 8*(lane>=16), N = lane&15) ----
#pragma unroll
  for (int r = 0; r < 8; ++r) {
    const int   m    = r + ((lane < 16) ? 0 : 8);
    const float inv  = rowinvS[m];
    const size_t orow = ((size_t)(b * N_ + i0 + m)) * F_;
    out[orow + wave * 32 + arow16]      = acc0[r] * inv;
    out[orow + wave * 32 + 16 + arow16] = acc1[r] * inv;
  }
}

// ---------------------------------------------------------------------------
extern "C" void kernel_launch(void* const* d_in, const int* in_sizes, int n_in,
                              void* d_out, int out_size, void* d_ws, size_t ws_size,
                              hipStream_t stream) {
  const float* word = (const float*)d_in[0];
  const float* adj  = (const float*)d_in[1];
  const float* W1   = (const float*)d_in[2];
  const float* W2   = (const float*)d_in[3];
  const float* w3   = (const float*)d_in[4];
  float* out = (float*)d_out;

  char*  ws = (char*)d_ws;
  float* v1 = (float*)ws;                    // 256 f32
  float* v2 = v1 + F_;                       // 256 f32
  float* s1 = v2 + F_;                       // B*N f32
  float* t2 = s1 + B_ * N_;                  // B*N f32
  float* s2 = t2 + B_ * N_;                  // B*N f32  (ends < 128 KB)
  __bf16* wT = (__bf16*)(ws + (128 << 10));  // 4 MB bf16 transposed word

  k_prep_v  <<<1,                 256, 0, stream>>>(W1, W2, w3, v1, v2);
  k_word_pre<<<B_ * (N_ / 32),    256, 0, stream>>>(word, v1, v2, s1, t2, wT);
  k_s2      <<<(B_ * N_) / 8,     256, 0, stream>>>(adj, t2, s2);
  k_attn_edge<<<B_ * (N_ / 16),   256, 0, stream>>>(adj, s1, s2, wT, out);
}
